// ExtendedEncoder_81758997446957
// MI455X (gfx1250) — matrix operand order repaired and checked by reading, not compiled, not run on hardware
//
#include <hip/hip_runtime.h>

// ---------------- problem constants (match reference) ----------------
#define LYR   6
#define DM    1024
#define HH    16
#define FF    4096
#define NB    33
#define DH    64
#define BB    4
#define SS    1024
#define ROWS  (BB * SS)          // 4096 rows of activations

typedef __attribute__((ext_vector_type(16))) __bf16 v16bf;
typedef __attribute__((ext_vector_type(8)))  float  v8f;
typedef __attribute__((ext_vector_type(4)))  int    v4i;

union FragAB { v4i i2[2]; v16bf b; };   // 32B: two b128 LDS chunks -> one WMMA operand

__device__ __forceinline__ float bf2f(unsigned short u) {
    union { unsigned int i; float f; } v; v.i = ((unsigned int)u) << 16; return v.f;
}
__device__ __forceinline__ unsigned short f2bf(float f) {
    union { float f; unsigned int i; } v; v.f = f;
    unsigned int u = v.i;
    u += 0x7FFFu + ((u >> 16) & 1u);   // round-to-nearest-even
    return (unsigned short)(u >> 16);
}
__device__ __forceinline__ v8f wmma_bf16(v16bf a, v16bf b, v8f c) {
    return __builtin_amdgcn_wmma_f32_16x16x32_bf16(false, a, false, b, (short)0, c, false, false);
}

// LDS offset of a __shared__ object: flat LDS addresses carry the wave-relative
// LDS offset in their low 32 bits (ISA 10.2 aperture rules).
__device__ __forceinline__ unsigned lds_off(const void* p) {
    return (unsigned)(unsigned long long)p;
}
// CDNA5 async DMA: global -> LDS, 16 bytes per lane, tracked by ASYNCcnt.
__device__ __forceinline__ void async_g2l_b128(unsigned dst_lds, const void* gsrc) {
    asm volatile("global_load_async_to_lds_b128 %0, %1, off"
                 :: "v"(dst_lds), "v"(gsrc)
                 : "memory");
}
__device__ __forceinline__ void wait_async0() {
    asm volatile("s_wait_asynccnt 0x0" ::: "memory");
}

// ---------------- elementwise kernels ----------------
__global__ void cvt_f32_bf16_kernel(const float* __restrict__ src,
                                    unsigned short* __restrict__ dst, int n) {
    int i = blockIdx.x * 256 + threadIdx.x;
    if (i < n) dst[i] = f2bf(src[i]);
}

__global__ void copy_f32_kernel(const float* __restrict__ src,
                                float* __restrict__ dst, int n) {
    int i = blockIdx.x * 256 + threadIdx.x;
    if (i < n) dst[i] = src[i];
}

// fp32 [K][N] -> bf16 transposed [N][K]; tiled for coalescing; K,N % 32 == 0.
// grid (N/32, K/32), block (32, 8).
__global__ void cvt_transpose_bf16_kernel(const float* __restrict__ src,
                                          unsigned short* __restrict__ dst,
                                          int K, int N) {
    __shared__ float tile[32][33];
    const int k0 = blockIdx.y * 32, n0 = blockIdx.x * 32;
    const int tx = threadIdx.x, ty = threadIdx.y;
#pragma unroll
    for (int i = 0; i < 32; i += 8)
        tile[ty + i][tx] = src[(size_t)(k0 + ty + i) * N + n0 + tx];
    __syncthreads();
#pragma unroll
    for (int i = 0; i < 32; i += 8)
        dst[(size_t)(n0 + ty + i) * K + k0 + tx] = f2bf(tile[tx][ty + i]);
}

// ---------------- generic bf16 WMMA GEMM (B pre-transposed) ----------------
// C[M,N] = A[M,K] @ B[K,N] + bias[N]; A is bf16 [M][K], Bt is bf16 [N][K].
// Block tile 128x128, 256 threads = 8 waves (2 x 4), each wave 64x32 (4x2 WMMA tiles).
// All staging is CDNA5 async DMA with LDS double buffering.
// M % 128 == 0, N % 128 == 0, K % 32 == 0 (true for all uses here).
__launch_bounds__(256)
__global__ void gemm_bf16_kernel(const unsigned short* __restrict__ A,
                                 const unsigned short* __restrict__ Bt,
                                 const float* __restrict__ bias,
                                 float* __restrict__ outF,
                                 unsigned short* __restrict__ outB,
                                 int M, int N, int K, int relu, int transOutB)
{
    __shared__ unsigned short Atile[2][128 * 32];   // [m][k], stride 32
    __shared__ unsigned short Btile[2][128 * 32];   // [n][k], stride 32

    const int t    = threadIdx.x;
    const int lane = t & 31;
    const int wave = t >> 5;
    const int m0   = blockIdx.y * 128;
    const int n0   = blockIdx.x * 128;
    const int wr   = (wave >> 2) * 64;   // wave row offset within block (0 / 64)
    const int wc   = (wave & 3) * 32;    // wave col offset within block (0..96)
    const int mrow = lane & 15;
    const int kg   = lane >> 4;          // low-K / high-K lane group

    // per-thread staging coordinates (two 16B chunks; same shape for A and Bt)
    const int r0 = (t * 8) >> 5,        c0 = (t * 8) & 31;
    const int r1 = (t * 8 + 2048) >> 5, c1 = (t * 8 + 2048) & 31;
    unsigned aL[2][2], bL[2][2];
#pragma unroll
    for (int buf = 0; buf < 2; ++buf) {
        aL[buf][0] = lds_off(&Atile[buf][r0 * 32 + c0]);
        aL[buf][1] = lds_off(&Atile[buf][r1 * 32 + c1]);
        bL[buf][0] = lds_off(&Btile[buf][r0 * 32 + c0]);
        bL[buf][1] = lds_off(&Btile[buf][r1 * 32 + c1]);
    }
    auto stage = [&](int buf, int kc) {
        async_g2l_b128(aL[buf][0], A  + (size_t)(m0 + r0) * K + kc + c0);
        async_g2l_b128(aL[buf][1], A  + (size_t)(m0 + r1) * K + kc + c1);
        async_g2l_b128(bL[buf][0], Bt + (size_t)(n0 + r0) * K + kc + c0);
        async_g2l_b128(bL[buf][1], Bt + (size_t)(n0 + r1) * K + kc + c1);
    };

    v8f zero = {0.f, 0.f, 0.f, 0.f, 0.f, 0.f, 0.f, 0.f};
    v8f acc[4][2];
#pragma unroll
    for (int tr = 0; tr < 4; ++tr)
#pragma unroll
        for (int tc = 0; tc < 2; ++tc) acc[tr][tc] = zero;

    stage(0, 0);
    wait_async0();
    __syncthreads();

    int cur = 0;
    for (int kc = 0; kc < K; kc += 32) {
        if (kc + 32 < K) stage(cur ^ 1, kc + 32);   // DMA next slab, overlap WMMA

        // fragments, contiguous k order (same for A and B):
        //  A: lane m (kg): row m, K kg*16..kg*16+15 contiguous
        //  B: lane n (kg): column n, K kg*16..kg*16+15 contiguous
        const unsigned short* At = Atile[cur];
        const unsigned short* Bb = Btile[cur];
        FragAB afrag[4], bfrag[2];
#pragma unroll
        for (int tr = 0; tr < 4; ++tr) {
            int m = wr + tr * 16 + mrow;
            afrag[tr].i2[0] = *(const v4i*)(&At[m * 32 + kg * 16]);
            afrag[tr].i2[1] = *(const v4i*)(&At[m * 32 + kg * 16 + 8]);
        }
#pragma unroll
        for (int tc = 0; tc < 2; ++tc) {
            int n = wc + tc * 16 + mrow;
            bfrag[tc].i2[0] = *(const v4i*)(&Bb[n * 32 + kg * 16]);
            bfrag[tc].i2[1] = *(const v4i*)(&Bb[n * 32 + kg * 16 + 8]);
        }
#pragma unroll
        for (int tr = 0; tr < 4; ++tr)
#pragma unroll
            for (int tc = 0; tc < 2; ++tc)
                acc[tr][tc] = wmma_bf16(afrag[tr].b, bfrag[tc].b, acc[tr][tc]);

        wait_async0();      // next slab landed
        __syncthreads();    // everyone done reading cur
        cur ^= 1;
    }

    // epilogue: D layout => VGPR r holds M = kg*8 + r, N = lane%16
#pragma unroll
    for (int tr = 0; tr < 4; ++tr) {
#pragma unroll
        for (int tc = 0; tc < 2; ++tc) {
#pragma unroll
            for (int r = 0; r < 8; ++r) {
                int m = m0 + wr + tr * 16 + kg * 8 + r;
                int n = n0 + wc + tc * 16 + mrow;
                float v = acc[tr][tc][r] + bias[n];
                if (relu) v = fmaxf(v, 0.f);
                if (outF) outF[(size_t)m * N + n] = v;
                if (outB) {
                    size_t idx = transOutB ? ((size_t)n * M + m) : ((size_t)m * N + n);
                    outB[idx] = f2bf(v);
                }
            }
        }
    }
}

// ---------------- flash-style relative-position attention ----------------
// grid: (S/64, H, B), 256 threads. q/k bf16 [ROWS][DM]; vT bf16 [DM][ROWS].
// scores = (q@k^T + qe_gather + brel_gather) * 0.125 ; online softmax ; O += P@V.
__launch_bounds__(256)
__global__ void attention_kernel(const unsigned short* __restrict__ qbf,
                                 const unsigned short* __restrict__ kbf,
                                 const unsigned short* __restrict__ vtbf, // [DM][ROWS]
                                 const unsigned short* __restrict__ erel, // [NB][DH] bf16
                                 const float* __restrict__ brel,          // [H][NB]
                                 const int* __restrict__ rel_ids,         // [S][S]
                                 unsigned short* __restrict__ abf)        // [ROWS][DM] bf16
{
    __shared__ unsigned short Qs[64 * 64];   // [q][d]
    __shared__ unsigned short KVs[64 * 64];  // K block [k][d]; then V^T [d][k]
    __shared__ float Ss[64 * 64];            // scores (fp32); P aliases it (see below)
    __shared__ float Os[64 * 64];            // output accumulator
    __shared__ float qe[64 * 36];            // qe[q][bucket] + brel folded in
    __shared__ float rowm[64], rowl[64];
    // P row q lives inside Ss row q's own 256B: Ps[q*128 + c] (per-thread-safe aliasing)
    unsigned short* Ps = (unsigned short*)Ss;

    const int t = threadIdx.x, lane = t & 31, wave = t >> 5;
    const int qt = blockIdx.x, h = blockIdx.y, b = blockIdx.z;
    const size_t rowbase = (size_t)b * SS;
    const int colh = h * DH;
    const float scale = 0.125f;  // 1/sqrt(64)

    v8f zero = {0.f, 0.f, 0.f, 0.f, 0.f, 0.f, 0.f, 0.f};

    // per-thread staging coordinates for 64x64 bf16 tiles (16B per thread per pass)
    const int tr0 = (t * 8) >> 6,        tc0 = (t * 8) & 63;
    const int tr1 = (t * 8 + 2048) >> 6, tc1 = (t * 8 + 2048) & 63;
    const unsigned kvLds0 = lds_off(&KVs[tr0 * 64 + tc0]);
    const unsigned kvLds1 = lds_off(&KVs[tr1 * 64 + tc1]);

    // load Q tile (64x64 bf16) via async DMA
    async_g2l_b128(lds_off(&Qs[tr0 * 64 + tc0]),
                   qbf + (rowbase + qt * 64 + tr0) * DM + colh + tc0);
    async_g2l_b128(lds_off(&Qs[tr1 * 64 + tc1]),
                   qbf + (rowbase + qt * 64 + tr1) * DM + colh + tc1);
    for (int i = t; i < 64 * 64; i += 256) Os[i] = 0.f;
    if (t < 64) { rowm[t] = -1e30f; rowl[t] = 0.f; }
    wait_async0();
    __syncthreads();

    // qe[q][nb] = dot(q_row, Erel[nb]) + Brel[h][nb]
    for (int idx = t; idx < 64 * NB; idx += 256) {
        int r = idx / NB, nb = idx % NB;
        float s = 0.f;
        for (int d = 0; d < DH; ++d) s += bf2f(Qs[r * 64 + d]) * bf2f(erel[nb * 64 + d]);
        qe[r * 36 + nb] = s + brel[h * NB + nb];
    }
    __syncthreads();

    const int wr = (wave >> 1) * 16;  // tile row base 0/16/32/48
    const int wc = (wave & 1) * 32;   // tile col base 0/32
    const int mrow = lane & 15, kg = lane >> 4;

    for (int kb = 0; kb < 16; ++kb) {
        // ph1: K block [k][d] (natural layout == B^T for S = Q K^T) via async DMA
        async_g2l_b128(kvLds0, kbf + (rowbase + kb * 64 + tr0) * DM + colh + tc0);
        async_g2l_b128(kvLds1, kbf + (rowbase + kb * 64 + tr1) * DM + colh + tc1);
        wait_async0();
        __syncthreads();

        // ph2: S tile via WMMA + bias gather + scale
#pragma unroll
        for (int tc = 0; tc < 2; ++tc) {
            v8f acc = zero;
#pragma unroll
            for (int ks = 0; ks < 2; ++ks) {
                FragAB af, bfg;
                int m = wr + mrow;
                af.i2[0] = *(const v4i*)(&Qs[m * 64 + ks * 32 + kg * 16]);
                af.i2[1] = *(const v4i*)(&Qs[m * 64 + ks * 32 + kg * 16 + 8]);
                int n = wc + tc * 16 + mrow;
                bfg.i2[0] = *(const v4i*)(&KVs[n * 64 + ks * 32 + kg * 16]);
                bfg.i2[1] = *(const v4i*)(&KVs[n * 64 + ks * 32 + kg * 16 + 8]);
                acc = wmma_bf16(af.b, bfg.b, acc);
            }
#pragma unroll
            for (int r = 0; r < 8; ++r) {
                int ql = wr + kg * 8 + r;
                int kl = wc + tc * 16 + mrow;
                int rid = rel_ids[(size_t)(qt * 64 + ql) * SS + kb * 64 + kl];
                Ss[ql * 64 + kl] = (acc[r] + qe[ql * 36 + rid]) * scale;
            }
        }
        __syncthreads();   // all K-fragment reads + Ss writes done

        // ph3: V^T block [d][k] via async DMA (vT is pre-transposed in global),
        // overlapped with ph4 softmax below.
        async_g2l_b128(kvLds0, vtbf + (size_t)(colh + tr0) * ROWS + rowbase + kb * 64 + tc0);
        async_g2l_b128(kvLds1, vtbf + (size_t)(colh + tr1) * ROWS + rowbase + kb * 64 + tc1);

        // ph4: online softmax per q row (thread t owns row t) -- overlaps V^T DMA
        if (t < 64) {
            float mold = rowm[t], mx = mold;
            for (int c = 0; c < 64; ++c) mx = fmaxf(mx, Ss[t * 64 + c]);
            float alpha = __expf(mold - mx);
            float s = 0.f;
            for (int c = 0; c < 64; ++c) {
                float p = __expf(Ss[t * 64 + c] - mx);
                s += p;
                Ps[t * 128 + c] = f2bf(p);   // writes stay inside row t's bytes
            }
            rowl[t] = rowl[t] * alpha + s;
            rowm[t] = mx;
            for (int c = 0; c < 64; ++c) Os[t * 64 + c] *= alpha;
        }
        wait_async0();
        __syncthreads();

        // ph5: O += P @ V  (A = P [q][k] stride 128, B^T = V^T [d][k] stride 64)
#pragma unroll
        for (int tc = 0; tc < 2; ++tc) {
            v8f acc = zero;
#pragma unroll
            for (int ks = 0; ks < 2; ++ks) {
                FragAB af, bfg;
                int m = wr + mrow;
                af.i2[0] = *(const v4i*)(&Ps[m * 128 + ks * 32 + kg * 16]);
                af.i2[1] = *(const v4i*)(&Ps[m * 128 + ks * 32 + kg * 16 + 8]);
                int n = wc + tc * 16 + mrow;
                bfg.i2[0] = *(const v4i*)(&KVs[n * 64 + ks * 32 + kg * 16]);
                bfg.i2[1] = *(const v4i*)(&KVs[n * 64 + ks * 32 + kg * 16 + 8]);
                acc = wmma_bf16(af.b, bfg.b, acc);
            }
#pragma unroll
            for (int r = 0; r < 8; ++r) {
                int ql = wr + kg * 8 + r;
                int dl = wc + tc * 16 + mrow;
                Os[ql * 64 + dl] += acc[r];
            }
        }
        __syncthreads();
    }

    // finalize: normalize, write bf16 'bqhd' layout (row = b*S+q, col = h*64+d)
    if (t < 64) {
        float inv = 1.f / rowl[t];
        for (int c = 0; c < 64; ++c)
            abf[(rowbase + qt * 64 + t) * DM + colh + c] = f2bf(Os[t * 64 + c] * inv);
    }
}

// ---------------- fused residual-add + LayerNorm ----------------
// out = LN(x + y) * g + be ; one block per row (D = 1024), fp32 + optional bf16 out.
__launch_bounds__(256)
__global__ void add_ln_kernel(const float* __restrict__ x, const float* __restrict__ y,
                              const float* __restrict__ g, const float* __restrict__ be,
                              float* __restrict__ outF, unsigned short* __restrict__ outB)
{
    __shared__ float sh[DM];
    __shared__ float red[256], red2[256];
    const int row = blockIdx.x, t = threadIdx.x;
    float s = 0.f, s2 = 0.f;
    for (int c = t; c < DM; c += 256) {
        float v = x[(size_t)row * DM + c] + y[(size_t)row * DM + c];
        sh[c] = v; s += v; s2 += v * v;
    }
    red[t] = s; red2[t] = s2;
    __syncthreads();
    for (int sft = 128; sft > 0; sft >>= 1) {
        if (t < sft) { red[t] += red[t + sft]; red2[t] += red2[t + sft]; }
        __syncthreads();
    }
    float mean = red[0] * (1.f / DM);
    float var  = red2[0] * (1.f / DM) - mean * mean;
    float rinv = rsqrtf(var + 1e-6f);
    for (int c = t; c < DM; c += 256) {
        float o = (sh[c] - mean) * rinv * g[c] + be[c];
        size_t idx = (size_t)row * DM + c;
        if (outF) outF[idx] = o;
        if (outB) outB[idx] = f2bf(o);
    }
}

// ---------------- host-side orchestration ----------------
static inline void launch_cvt(const float* s, unsigned short* d, int n, hipStream_t st) {
    cvt_f32_bf16_kernel<<<(n + 255) / 256, 256, 0, st>>>(s, d, n);
}
static inline void launch_cvt_T(const float* s, unsigned short* d, int K, int N,
                                hipStream_t st) {
    cvt_transpose_bf16_kernel<<<dim3(N / 32, K / 32), dim3(32, 8), 0, st>>>(s, d, K, N);
}

extern "C" void kernel_launch(void* const* d_in, const int* in_sizes, int n_in,
                              void* d_out, int out_size, void* d_ws, size_t ws_size,
                              hipStream_t stream) {
    const float* x    = (const float*)d_in[0];
    const int*   rel  = (const int*)  d_in[1];
    const float* Wq   = (const float*)d_in[2];
    const float* bq   = (const float*)d_in[3];
    const float* Wk   = (const float*)d_in[4];
    const float* bk   = (const float*)d_in[5];
    const float* Wv   = (const float*)d_in[6];
    const float* bv   = (const float*)d_in[7];
    const float* Wo   = (const float*)d_in[8];
    const float* bo   = (const float*)d_in[9];
    const float* Erel = (const float*)d_in[10];
    const float* Brel = (const float*)d_in[11];
    const float* g1   = (const float*)d_in[12];
    const float* be1  = (const float*)d_in[13];
    const float* g2   = (const float*)d_in[14];
    const float* be2  = (const float*)d_in[15];
    const float* W1   = (const float*)d_in[16];
    const float* b1   = (const float*)d_in[17];
    const float* W2   = (const float*)d_in[18];
    const float* b2   = (const float*)d_in[19];

    float* xa = (float*)d_out;   // fp32 activation buffer == final output

    char* base = (char*)d_ws;
    size_t off = 0;
    auto take = [&](size_t bytes) -> char* {
        char* p = base + off;
        off = (off + bytes + 255) & ~(size_t)255;
        return p;
    };
    // transposed bf16 weights (Bt layout [N][K])
    unsigned short* wqb   = (unsigned short*)take((size_t)DM * DM * 2);
    unsigned short* wkb   = (unsigned short*)take((size_t)DM * DM * 2);
    unsigned short* wvb   = (unsigned short*)take((size_t)DM * DM * 2);
    unsigned short* wob   = (unsigned short*)take((size_t)DM * DM * 2);
    unsigned short* w1b   = (unsigned short*)take((size_t)DM * FF * 2);
    unsigned short* w2b   = (unsigned short*)take((size_t)FF * DM * 2);
    unsigned short* erelb = (unsigned short*)take((size_t)NB * DH * 2);
    unsigned short* xbf   = (unsigned short*)take((size_t)ROWS * DM * 2);
    unsigned short* qb    = (unsigned short*)take((size_t)ROWS * DM * 2);
    unsigned short* kb    = (unsigned short*)take((size_t)ROWS * DM * 2);
    unsigned short* vtb   = (unsigned short*)take((size_t)DM * ROWS * 2);  // V^T
    unsigned short* ab    = (unsigned short*)take((size_t)ROWS * DM * 2);
    float*          proj  = (float*)         take((size_t)ROWS * DM * 4);
    float*          ffin  = (float*)         take((size_t)ROWS * DM * 4);
    unsigned short* ffinb = (unsigned short*)take((size_t)ROWS * DM * 2);
    unsigned short* h1b   = (unsigned short*)take((size_t)ROWS * FF * 2);
    (void)ws_size; (void)in_sizes; (void)n_in; (void)out_size;

    // x -> activation buffer
    copy_f32_kernel<<<(ROWS * DM + 255) / 256, 256, 0, stream>>>(x, xa, ROWS * DM);

    const dim3 gemmBlk(256);
    const dim3 gridDD(DM / 128, ROWS / 128);    // N=1024
    const dim3 gridDF(FF / 128, ROWS / 128);    // N=4096
    const dim3 attnGrid(SS / 64, HH, BB);
    const dim3 lnGrid(ROWS);

    for (int l = 0; l < LYR; ++l) {
        const size_t oDD = (size_t)l * DM * DM;
        const size_t oDF = (size_t)l * DM * FF;
        // per-layer weight conversion to transposed bf16 [N][K]
        launch_cvt_T(Wq + oDD, wqb, DM, DM, stream);
        launch_cvt_T(Wk + oDD, wkb, DM, DM, stream);
        launch_cvt_T(Wv + oDD, wvb, DM, DM, stream);
        launch_cvt_T(Wo + oDD, wob, DM, DM, stream);
        launch_cvt_T(W1 + oDF, w1b, DM, FF, stream);   // [FF][DM]
        launch_cvt_T(W2 + oDF, w2b, FF, DM, stream);   // [DM][FF]
        launch_cvt(Erel + (size_t)l * NB * DH, erelb, NB * DH, stream);
        // current activation -> bf16
        launch_cvt(xa, xbf, ROWS * DM, stream);

        // QKV projections (bf16 out; V written transposed for attention DMA)
        gemm_bf16_kernel<<<gridDD, gemmBlk, 0, stream>>>(xbf, wqb, bq + (size_t)l * DM,
                                                         nullptr, qb, ROWS, DM, DM, 0, 0);
        gemm_bf16_kernel<<<gridDD, gemmBlk, 0, stream>>>(xbf, wkb, bk + (size_t)l * DM,
                                                         nullptr, kb, ROWS, DM, DM, 0, 0);
        gemm_bf16_kernel<<<gridDD, gemmBlk, 0, stream>>>(xbf, wvb, bv + (size_t)l * DM,
                                                         nullptr, vtb, ROWS, DM, DM, 0, 1);
        // attention
        attention_kernel<<<attnGrid, gemmBlk, 0, stream>>>(
            qb, kb, vtb, erelb, Brel + (size_t)l * HH * NB, rel, ab);
        // output projection (fp32 out)
        gemm_bf16_kernel<<<gridDD, gemmBlk, 0, stream>>>(ab, wob, bo + (size_t)l * DM,
                                                         proj, nullptr, ROWS, DM, DM, 0, 0);
        // ff_in = LN(x + attn_out)
        add_ln_kernel<<<lnGrid, gemmBlk, 0, stream>>>(xa, proj, g1 + (size_t)l * DM,
                                                      be1 + (size_t)l * DM, ffin, ffinb);
        // FFN
        gemm_bf16_kernel<<<gridDF, gemmBlk, 0, stream>>>(ffinb, w1b, b1 + (size_t)l * FF,
                                                         nullptr, h1b, ROWS, FF, DM, 1, 0);
        gemm_bf16_kernel<<<gridDD, gemmBlk, 0, stream>>>(h1b, w2b, b2 + (size_t)l * DM,
                                                         proj, nullptr, ROWS, DM, FF, 0, 0);
        // x = LN(ff_in + ff)  -> activation buffer (== d_out on last layer)
        add_ln_kernel<<<lnGrid, gemmBlk, 0, stream>>>(ffin, proj, g2 + (size_t)l * DM,
                                                      be2 + (size_t)l * DM, xa, nullptr);
    }
}
